// AttentionBlock3d_1898375544865
// MI455X (gfx1250) — matrix operand order repaired and checked by reading, not compiled, not run on hardware
//
#include <hip/hip_runtime.h>
#include <hip/hip_bf16.h>

// ---------------------------------------------------------------------------
// AttentionBlock3d for MI455X (gfx1250, wave32, WMMA bf16 16x16x32).
// B=1, C=64, N=4608 (8*24*24), 4 heads, head_dim=16.
// k1: qkv projection  -> Q,K bf16 [head][n][16] (Q pre-scaled by 0.25*log2e),
//                        V bf16 [head][16][N]
// k2: flash attention -> h f32 [64][N]
//     - double-buffered K/V tiles staged with GLOBAL_LOAD_ASYNC_TO_LDS_B128
//     - softmax in log2 domain; row reductions via DPP16 row_xmask butterflies
//     - native v_cvt bf16 conversions (hardware cast, not bit-twiddled)
// k3: out projection + bias + residual -> out f32 [64][N]
// ---------------------------------------------------------------------------

#define NSEQ   4608
#define NHEADS 4
#define HD     16
#define CCH    64
#define NJT    (NSEQ / 64)

typedef __attribute__((ext_vector_type(4)))  unsigned int u32x4;
typedef __attribute__((ext_vector_type(16))) __bf16       v16bf;
typedef __attribute__((ext_vector_type(8)))  float        v8f;

struct Frag { u32x4 q[2]; };                       // 32 bytes = 8 VGPRs
static __device__ inline v16bf asbf(const Frag& f) {
    return __builtin_bit_cast(v16bf, f);
}
static __device__ inline unsigned short f2bf(float f) {
    return __builtin_bit_cast(unsigned short, (__bf16)f);  // native v_cvt
}

// DPP16 row_xmask lane swap within a 16-lane row: dpp_ctrl = 0x160 | mask.
template <int MASK>
static __device__ inline float dpp_xor16(float v) {
    return __builtin_bit_cast(float,
        __builtin_amdgcn_update_dpp(0, __builtin_bit_cast(int, v),
                                    0x160 | MASK, 0xf, 0xf, true));
}
static __device__ inline float row_max16(float t) {
    t = fmaxf(t, dpp_xor16<1>(t));
    t = fmaxf(t, dpp_xor16<2>(t));
    t = fmaxf(t, dpp_xor16<4>(t));
    t = fmaxf(t, dpp_xor16<8>(t));
    return t;
}
static __device__ inline float row_sum16(float t) {
    t += dpp_xor16<1>(t);
    t += dpp_xor16<2>(t);
    t += dpp_xor16<4>(t);
    t += dpp_xor16<8>(t);
    return t;
}

// ---------------------------------------------------------------------------
// Kernel 1: qkv = w_qkv(192x64) @ x(64xN); scatter into q/k/v bf16 layouts.
// Q pre-scaled by (1/sqrt(hd)) * log2(e) so softmax runs in the log2 domain.
// ---------------------------------------------------------------------------
__global__ __launch_bounds__(256) void qkv_kernel(
    const float* __restrict__ x, const float* __restrict__ wqkv,
    unsigned short* __restrict__ qws, unsigned short* __restrict__ kws,
    unsigned short* __restrict__ vws)
{
    const int n = blockIdx.x * 256 + threadIdx.x;      // < 4608 exactly
    const float qscale = 0.25f * 1.44269504088896341f; // 1/sqrt(16) * log2(e)
    float xr[CCH];
#pragma unroll
    for (int c = 0; c < CCH; ++c) xr[c] = x[c * NSEQ + n];

    for (int hh = 0; hh < NHEADS; ++hh) {
        for (int cp = 0; cp < HD; ++cp) {
            const int oq = hh * 48 + cp;               // qkv row for q
            float aq = 0.f, ak = 0.f, av = 0.f;
#pragma unroll
            for (int c = 0; c < CCH; ++c) {
                aq = fmaf(wqkv[(oq)      * CCH + c], xr[c], aq);
                ak = fmaf(wqkv[(oq + 16) * CCH + c], xr[c], ak);
                av = fmaf(wqkv[(oq + 32) * CCH + c], xr[c], av);
            }
            qws[((size_t)(hh * NSEQ + n)) * HD + cp] = f2bf(aq * qscale);
            kws[((size_t)(hh * NSEQ + n)) * HD + cp] = f2bf(ak);
            vws[((size_t)(hh * HD + cp)) * NSEQ + n] = f2bf(av);
        }
    }
}

// ---------------------------------------------------------------------------
// Kernel 2: flash attention. grid = NHEADS * (NSEQ/64) blocks, 128 thr = 4 waves.
// ---------------------------------------------------------------------------
__global__ __launch_bounds__(128) void attn_kernel(
    const unsigned short* __restrict__ qws, const unsigned short* __restrict__ kws,
    const unsigned short* __restrict__ vws, float* __restrict__ hws)
{
    __shared__ __align__(16) unsigned short sK[2][64 * HD];   // [buf][j][c] 4KB
    __shared__ __align__(16) unsigned short sV[2][HD * 64];   // [buf][c][j] 4KB
    __shared__ __align__(16) unsigned short sP[4][16 * 64];   // per-wave P  8KB

    const int head  = blockIdx.x / NJT;
    const int i0    = (blockIdx.x % NJT) * 64;
    const int wave  = threadIdx.x >> 5;
    const int lane  = threadIdx.x & 31;
    const int col   = lane & 15;          // C/D column, B column, A row
    const int hi    = lane >> 4;          // which half-wave
    const int rbase = hi * 8;             // C/D rows this lane holds
    const int iw0   = i0 + wave * 16;

    // Async tile staging addressing:
    // K tile: 64*16 bf16 = 2KB contiguous; 128 threads x 16B.
    // V tile: 16 rows x 128B; thread t -> row t/8, 16B chunk t%8.
    const int vc = threadIdx.x >> 3, vp = threadIdx.x & 7;
    const unsigned long long gK =
        (unsigned long long)(uintptr_t)(kws + (size_t)head * NSEQ * HD);
    const unsigned long long gV =
        (unsigned long long)(uintptr_t)(vws + (size_t)head * HD * NSEQ);
    const unsigned ldsKoff = threadIdx.x * 16u;                 // bytes in K tile
    const unsigned ldsVoff = (unsigned)(vc * 64 + vp * 8) * 2u; // bytes in V tile
    const unsigned gVoff   = (unsigned)(vc * NSEQ + vp * 8) * 2u;

    auto issue_tile = [&](int j0, int buf) {
        const unsigned dK = (unsigned)(uintptr_t)(&sK[buf][0]) + ldsKoff;
        const unsigned oK = (unsigned)(j0 * HD * 2) + ldsKoff;
        asm volatile("global_load_async_to_lds_b128 %0, %1, %2"
                     :: "v"(dK), "v"(oK), "s"(gK) : "memory");
        const unsigned dV = (unsigned)(uintptr_t)(&sV[buf][0]) + ldsVoff;
        const unsigned oV = (unsigned)(j0 * 2) + gVoff;
        asm volatile("global_load_async_to_lds_b128 %0, %1, %2"
                     :: "v"(dV), "v"(oV), "s"(gV) : "memory");
    };

    // --- Q A-fragment (16x32, K=c padded 16->32 with zeros), loaded once ---
    Frag aq;
    {
        const u32x4 z = {0u, 0u, 0u, 0u};
        const u32x4* src = (const u32x4*)(qws +
            ((size_t)(head * NSEQ + iw0 + col)) * HD + hi * 8);
        aq.q[0] = src[0];      // K 0..7 (hi=0) or 8..15 (hi=1)
        aq.q[1] = z;           // K 16..31 = padding
    }

    float mrow[8], lrow[8];
#pragma unroll
    for (int r = 0; r < 8; ++r) { mrow[r] = -1e30f; lrow[r] = 0.f; }
    v8f hacc = {};

    issue_tile(0, 0);                                  // prologue prefetch

    for (int jt = 0; jt < NJT; ++jt) {
        const int buf = jt & 1;
        const unsigned short* sKb = sK[buf];
        const unsigned short* sVb = sV[buf];

        if (jt + 1 < NJT) {
            issue_tile((jt + 1) * 64, buf ^ 1);        // prefetch next tile
            asm volatile("s_wait_asynccnt 0x2" ::: "memory"); // current tile done
        } else {
            asm volatile("s_wait_asynccnt 0x0" ::: "memory");
        }
        __syncthreads();                               // tile visible to all waves

        // ---- S = Q^T K : 4 subtiles of 16 j-columns ----
        v8f sacc[4];
#pragma unroll
        for (int jj = 0; jj < 4; ++jj) {
            Frag bk;
            const u32x4 z = {0u, 0u, 0u, 0u};
            {                              // K(c)=hi*8..hi*8+7 of column j; K 16..31 -> padding
                const u32x4* p = (const u32x4*)(sKb + (jj * 16 + col) * HD + hi * 8);
                bk.q[0] = p[0]; bk.q[1] = z;
            }
            v8f cz = {};
            sacc[jj] = __builtin_amdgcn_wmma_f32_16x16x32_bf16(
                false, asbf(aq), false, asbf(bk), (short)0, cz, false, false);
        }

        // ---- online softmax (log2 domain; scale folded into Q) ----
        float alpha[8], mnew[8], rs[8], pv[4][8];
#pragma unroll
        for (int r = 0; r < 8; ++r) {
            float t = fmaxf(fmaxf(sacc[0][r], sacc[1][r]),
                            fmaxf(sacc[2][r], sacc[3][r]));
            t = row_max16(t);                           // DPP16 xor butterfly
            mnew[r]  = fmaxf(mrow[r], t);
            alpha[r] = __builtin_amdgcn_exp2f(mrow[r] - mnew[r]);
            mrow[r]  = mnew[r];
        }
#pragma unroll
        for (int r = 0; r < 8; ++r) {
            float s = 0.f;
#pragma unroll
            for (int jj = 0; jj < 4; ++jj) {
                const float p = __builtin_amdgcn_exp2f(sacc[jj][r] - mnew[r]);
                pv[jj][r] = p;
                s += p;
            }
            rs[r] = row_sum16(s);                       // DPP16 xor butterfly
        }
#pragma unroll
        for (int r = 0; r < 8; ++r) {
            lrow[r] = lrow[r] * alpha[r] + rs[r];
            hacc[r] *= alpha[r];
        }

        // ---- P (C-layout) -> LDS -> A-fragment layout (wave-private strip) ----
        unsigned short* pb = sP[wave];
#pragma unroll
        for (int jj = 0; jj < 4; ++jj)
#pragma unroll
            for (int r = 0; r < 8; ++r)
                pb[(rbase + r) * 64 + jj * 16 + col] = f2bf(pv[jj][r]);

        // ---- O += P.V : two K=32 WMMAs cover the 64-wide j-tile ----
#pragma unroll
        for (int kk = 0; kk < 2; ++kk) {
            Frag pa, vb;
            pa.q[0] = *((const u32x4*)(pb + col * 64 + kk * 32 + hi * 8));
            pa.q[1] = *((const u32x4*)(pb + col * 64 + kk * 32 + 16 + hi * 8));
            const u32x4* pv2 = (const u32x4*)(sVb + col * 64 + kk * 32 + hi * 8);
            vb.q[0] = pv2[0]; vb.q[1] = pv2[2];
            hacc = __builtin_amdgcn_wmma_f32_16x16x32_bf16(
                false, asbf(pa), false, asbf(vb), (short)0, hacc, false, false);
        }
        __syncthreads();   // all waves done with this buffer before it is refilled
    }

    // ---- finalize: h[c][i] = O / l ----
#pragma unroll
    for (int r = 0; r < 8; ++r) {
        const int i = iw0 + rbase + r;
        hws[(size_t)(head * HD + col) * NSEQ + i] = hacc[r] / lrow[r];
    }
}

// ---------------------------------------------------------------------------
// Kernel 3: out = w_proj(64x64) @ h + b + x  (bandwidth bound)
// ---------------------------------------------------------------------------
__global__ __launch_bounds__(256) void proj_kernel(
    const float* __restrict__ h, const float* __restrict__ wproj,
    const float* __restrict__ bproj, const float* __restrict__ x,
    float* __restrict__ out)
{
    const int n = blockIdx.x * 256 + threadIdx.x;
    float hr[CCH];
#pragma unroll
    for (int c = 0; c < CCH; ++c) hr[c] = h[c * NSEQ + n];

    for (int o = 0; o < CCH; ++o) {
        float acc = bproj[o];
#pragma unroll
        for (int c = 0; c < CCH; ++c) acc = fmaf(wproj[o * CCH + c], hr[c], acc);
        out[o * NSEQ + n] = acc + x[o * NSEQ + n];
    }
}

// ---------------------------------------------------------------------------
extern "C" void kernel_launch(void* const* d_in, const int* in_sizes, int n_in,
                              void* d_out, int out_size, void* d_ws, size_t ws_size,
                              hipStream_t stream)
{
    const float* x     = (const float*)d_in[0];
    const float* wqkv  = (const float*)d_in[1];
    const float* wproj = (const float*)d_in[2];
    const float* bproj = (const float*)d_in[3];
    float* out = (float*)d_out;

    unsigned short* qws = (unsigned short*)d_ws;
    unsigned short* kws = qws + (size_t)NHEADS * NSEQ * HD;
    unsigned short* vws = kws + (size_t)NHEADS * NSEQ * HD;
    float*          hws = (float*)(vws + (size_t)NHEADS * HD * NSEQ);

    qkv_kernel <<<NSEQ / 256, 256, 0, stream>>>(x, wqkv, qws, kws, vws);
    attn_kernel<<<NHEADS * NJT, 128, 0, stream>>>(qws, kws, vws, hws);
    proj_kernel<<<NSEQ / 256, 256, 0, stream>>>(hws, wproj, bproj, x, out);
}
